// ContrastiveLoss_74113955659998
// MI455X (gfx1250) — compile-verified, matrix operand order checked by reading, not run on hardware
//
#include <hip/hip_runtime.h>
#include <hip/hip_bf16.h>
#include <math.h>

typedef _Float16 v16h __attribute__((ext_vector_type(16)));
typedef _Float16 v8h  __attribute__((ext_vector_type(8)));
typedef _Float16 v4h  __attribute__((ext_vector_type(4)));
typedef float    v8f  __attribute__((ext_vector_type(8)));

#define D_DIM 128

// ---------------------------------------------------------------------------
// Kernel 1: L2-normalize rows (F.normalize, eps=1e-12), convert to f16, zero
// the per-row accumulators. One wave per row; lane owns 4 consecutive floats.
// ---------------------------------------------------------------------------
__global__ __launch_bounds__(256)
void cl_normalize_kernel(const float* __restrict__ emb,
                         _Float16* __restrict__ z16,
                         float* __restrict__ sum_all,
                         float* __restrict__ sum_pos,
                         int B) {
  const int wave = threadIdx.x >> 5;
  const int lane = threadIdx.x & 31;
  const int row  = blockIdx.x * 8 + wave;
  if (row >= B) return;

  const float4 v = ((const float4*)(emb + (size_t)row * D_DIM))[lane];
  float s = v.x * v.x + v.y * v.y + v.z * v.z + v.w * v.w;
#pragma unroll
  for (int msk = 16; msk >= 1; msk >>= 1) s += __shfl_xor(s, msk, 32);
  const float inv = 1.0f / fmaxf(sqrtf(s), 1e-12f);

  v4h h;
  h[0] = (_Float16)(v.x * inv);
  h[1] = (_Float16)(v.y * inv);
  h[2] = (_Float16)(v.z * inv);
  h[3] = (_Float16)(v.w * inv);
  ((v4h*)(z16 + (size_t)row * D_DIM))[lane] = h;

  if (lane == 0) {
    sum_all[row] = 0.0f;
    sum_pos[row] = 0.0f;
  }
}

// ---------------------------------------------------------------------------
// 16-bit 16x32 operand fragment per ISA 7.12.2 (A rows / B columns identical
// pattern for sim = Z Z^T). Two aligned 16-byte loads per lane.
// ---------------------------------------------------------------------------
__device__ __forceinline__ v16h load_frag(const _Float16* __restrict__ rowp,
                                          int kc, int hs) {
  const v8h lo = *(const v8h*)(rowp + kc * 32 + hs * 8);
  const v8h hi = *(const v8h*)(rowp + kc * 32 + 16 + hs * 8);
  return __builtin_shufflevector(lo, hi, 0, 1, 2, 3, 4, 5, 6, 7,
                                 8, 9, 10, 11, 12, 13, 14, 15);
}

// Branchless per-tile epilogue: e is an unconditional operand of the adds, so
// clang cannot sink the exp under EXEC-masked branches. Positive mask is one
// unsigned range test (censor pre-folded into tb). Diagonal handled elsewhere.
__device__ __forceinline__ void epi_tile(const v8f& cc, const int (&tb)[8],
                                         float (&aA)[8], float (&aP)[8],
                                         int tj) {
#pragma unroll
  for (int r = 0; r < 8; ++r) {
    const float e = __expf(fmaf(10.0f, cc[r], -10.0f));  // shift-by-max trick
    const unsigned td = (unsigned)(tb[r] - tj);          // |dt|<365 && censor
    const float mt = (td < 729u) ? 1.0f : 0.0f;
    aA[r] += e;
    aP[r] = fmaf(e, mt, aP[r]);
  }
}

// Wave-uniform diagonal correction: executed for the single j-tile whose base
// equals this i-tile's base. Subtract the diag term from both accumulators.
__device__ __forceinline__ void fix_diag(const v8f& cc, const int (&tb)[8],
                                         float (&aA)[8], float (&aP)[8],
                                         int tj, int hs, int m) {
#pragma unroll
  for (int r = 0; r < 8; ++r) {
    const float md = (((hs << 3) + r) == m) ? 1.0f : 0.0f;
    const float e  = __expf(fmaf(10.0f, cc[r], -10.0f)) * md;
    const unsigned td = (unsigned)(tb[r] - tj);
    const float mt = (td < 729u) ? 1.0f : 0.0f;
    aA[r] -= e;
    aP[r] = fmaf(-e, mt, aP[r]);
  }
}

// ---------------------------------------------------------------------------
// Kernel 2: fused implicit-GEMM + masked exp-sum epilogue.
// Block = 256 threads = 8 waves; each wave owns TWO 16-row i-tiles (32 rows)
// so every B fragment feeds 2 WMMAs. blockIdx.y splits the j dimension; the
// exp-sums are additive across j and are committed with float atomics.
// ---------------------------------------------------------------------------
__global__ __launch_bounds__(256)
void cl_sim_lse_kernel(const _Float16* __restrict__ z16,
                       const int* __restrict__ tsurv,
                       const int* __restrict__ cens,
                       float* __restrict__ sum_all,
                       float* __restrict__ sum_pos,
                       int B) {
  const int lane = threadIdx.x & 31;
  const int wave = threadIdx.x >> 5;
  const int m    = lane & 15;
  const int hs   = lane >> 4;
  const int ibase = blockIdx.x * 256 + wave * 32;

  // A fragments for the wave's 32 rows, K=0..127, resident in VGPRs.
  v16h a[2][4];
#pragma unroll
  for (int t = 0; t < 2; ++t) {
    const _Float16* arow = z16 + (size_t)(ibase + t * 16 + m) * D_DIM;
#pragma unroll
    for (int kc = 0; kc < 4; ++kc) a[t][kc] = load_frag(arow, kc, hs);
  }

  // Pre-folded positive-pair bound: td = tb - tj in [0,729) iff
  // (censor_i==1 && |t_i - t_j| < 365). Sentinel makes it always-false.
  int tb[2][8];
#pragma unroll
  for (int t = 0; t < 2; ++t)
#pragma unroll
    for (int r = 0; r < 8; ++r) {
      const int gi = ibase + t * 16 + hs * 8 + r;
      tb[t][r] = (cens[gi] == 1) ? (tsurv[gi] + 364) : 0x7FFF0000;
    }

  float accA[2][8], accP[2][8];
#pragma unroll
  for (int t = 0; t < 2; ++t)
#pragma unroll
    for (int r = 0; r < 8; ++r) { accA[t][r] = 0.0f; accP[t][r] = 0.0f; }

  const int jtpb = (B >> 4) / gridDim.y;
  const int jt0  = blockIdx.y * jtpb;

  for (int jt = 0; jt < jtpb; ++jt) {
    const int jbase = (jt0 + jt) << 4;
    const _Float16* brow = z16 + (size_t)(jbase + m) * D_DIM;

    if (jt + 1 < jtpb) __builtin_prefetch(brow + D_DIM * 16, 0, 3);

    v8f c0 = {0.f, 0.f, 0.f, 0.f, 0.f, 0.f, 0.f, 0.f};
    v8f c1 = {0.f, 0.f, 0.f, 0.f, 0.f, 0.f, 0.f, 0.f};
#pragma unroll
    for (int kc = 0; kc < 4; ++kc) {
      const v16h b = load_frag(brow, kc, hs);
      c0 = __builtin_amdgcn_wmma_f32_16x16x32_f16(false, a[0][kc], false, b,
                                                  (short)0, c0, false, false);
      c1 = __builtin_amdgcn_wmma_f32_16x16x32_f16(false, a[1][kc], false, b,
                                                  (short)0, c1, false, false);
    }

    const int tj = tsurv[jbase + m];
    epi_tile(c0, tb[0], accA[0], accP[0], tj);
    epi_tile(c1, tb[1], accA[1], accP[1], tj);

    // Diagonal lives in exactly one j-tile per i-tile: uniform branches.
    if (jbase == ibase)      fix_diag(c0, tb[0], accA[0], accP[0], tj, hs, m);
    if (jbase == ibase + 16) fix_diag(c1, tb[1], accA[1], accP[1], tj, hs, m);
  }

  // Reduce across the 16 lanes of each half-wave, then commit with atomics.
#pragma unroll
  for (int t = 0; t < 2; ++t)
#pragma unroll
    for (int r = 0; r < 8; ++r) {
      float sa = accA[t][r], sp = accP[t][r];
#pragma unroll
      for (int msk = 8; msk >= 1; msk >>= 1) {
        sa += __shfl_xor(sa, msk, 32);
        sp += __shfl_xor(sp, msk, 32);
      }
      if (m == 0) {
        const int gi = ibase + t * 16 + hs * 8 + r;
        atomicAdd(&sum_all[gi], sa);
        atomicAdd(&sum_pos[gi], sp);
      }
    }
}

// ---------------------------------------------------------------------------
// Kernel 3: per_row = log(sum_all) - log(sum_pos) when sum_pos>0 (exactly iff
// the row has a positive, since every contributing term >= e^-20 > 0);
// loss = sum / cnt (or 0).
// ---------------------------------------------------------------------------
__global__ __launch_bounds__(256)
void cl_finalize_kernel(const float* __restrict__ sum_all,
                        const float* __restrict__ sum_pos,
                        float* __restrict__ out, int B) {
  __shared__ float s_sum[256];
  __shared__ float s_cnt[256];
  float ls = 0.0f, lc = 0.0f;
  for (int i = threadIdx.x; i < B; i += 256) {
    const float sp = sum_pos[i];
    if (sp > 0.0f) {
      ls += logf(sum_all[i]) - logf(sp);
      lc += 1.0f;
    }
  }
  s_sum[threadIdx.x] = ls;
  s_cnt[threadIdx.x] = lc;
  __syncthreads();
  for (int s = 128; s > 0; s >>= 1) {
    if (threadIdx.x < (unsigned)s) {
      s_sum[threadIdx.x] += s_sum[threadIdx.x + s];
      s_cnt[threadIdx.x] += s_cnt[threadIdx.x + s];
    }
    __syncthreads();
  }
  if (threadIdx.x == 0)
    out[0] = (s_cnt[0] > 0.0f) ? (s_sum[0] / s_cnt[0]) : 0.0f;
}

// ---------------------------------------------------------------------------
extern "C" void kernel_launch(void* const* d_in, const int* in_sizes, int n_in,
                              void* d_out, int out_size, void* d_ws, size_t ws_size,
                              hipStream_t stream) {
  const float* emb   = (const float*)d_in[0];
  const int*   tsurv = (const int*)d_in[1];
  const int*   cens  = (const int*)d_in[2];
  float*       out   = (float*)d_out;

  const int B = in_sizes[1];  // 8192

  char* ws = (char*)d_ws;
  _Float16* z16     = (_Float16*)ws;
  float*    sum_all = (float*)(ws + (size_t)B * D_DIM * sizeof(_Float16));
  float*    sum_pos = sum_all + B;

  cl_normalize_kernel<<<dim3((B + 7) / 8), dim3(256), 0, stream>>>(
      emb, z16, sum_all, sum_pos, B);

  // 256 i-rows per block; j split 16 ways (exp-sums are additive over j).
  const int JSPLIT = 16;
  cl_sim_lse_kernel<<<dim3(B / 256, JSPLIT), dim3(256), 0, stream>>>(
      z16, tsurv, cens, sum_all, sum_pos, B);

  cl_finalize_kernel<<<dim3(1), dim3(256), 0, stream>>>(sum_all, sum_pos, out, B);
}
